// Self_Attn_12661563588949
// MI455X (gfx1250) — matrix operand order repaired and checked by reading, not compiled, not run on hardware
//
#include <hip/hip_runtime.h>

#define BB 4
#define TT 4
#define WD 40
#define HD 40
#define CIN 256
#define CS 32
#define PP (WD*HD)          // 1600 spatial positions per frame
#define NN (TT*PP)          // 6400 tokens per batch element
#define KT 32               // keys per tile
#define NKT (NN/KT)         // 200 key tiles

typedef __attribute__((ext_vector_type(16))) __bf16 v16bf;
typedef __attribute__((ext_vector_type(8)))  float  v8f;
typedef __attribute__((ext_vector_type(2)))  int    v2i;
typedef __attribute__((address_space(1))) v2i* gv2i_p;   // global (device) v2i*
typedef __attribute__((address_space(3))) v2i* lv2i_p;   // LDS (shared) v2i*

union AB { unsigned int u[8]; v16bf v; };

__device__ __forceinline__ unsigned short f2bf(float f) {
  unsigned int u = __float_as_uint(f);
  u += 0x7FFFu + ((u >> 16) & 1u);          // round-to-nearest-even
  return (unsigned short)(u >> 16);
}
__device__ __forceinline__ unsigned int pk2(unsigned short lo, unsigned short hi) {
  return (unsigned int)lo | ((unsigned int)hi << 16);
}

// ---- DPP16 butterfly reductions over a 16-lane row (VALU, no LDS) ----------
// 0xB1 = quad_perm(1,0,3,2)  -> xor 1
// 0x4E = quad_perm(2,3,0,1)  -> xor 2
// 0x141 = row_half_mirror    -> i^7 within 8 lanes
// 0x140 = row_mirror         -> i^15 within 16 lanes
template <int CTRL>
__device__ __forceinline__ float dppf(float x) {
  return __int_as_float(__builtin_amdgcn_update_dpp(
      0, __float_as_int(x), CTRL, 0xF, 0xF, true));
}
__device__ __forceinline__ float rowmax16(float v) {
  v = fmaxf(v, dppf<0xB1>(v));
  v = fmaxf(v, dppf<0x4E>(v));
  v = fmaxf(v, dppf<0x141>(v));
  v = fmaxf(v, dppf<0x140>(v));
  return v;
}
__device__ __forceinline__ float rowsum16(float v) {
  v += dppf<0xB1>(v);
  v += dppf<0x4E>(v);
  v += dppf<0x141>(v);
  v += dppf<0x140>(v);
  return v;
}

#if __has_builtin(__builtin_amdgcn_global_load_async_to_lds_b64)
#define HAS_ASYNC_LDS 1
#else
#define HAS_ASYNC_LDS 0
#endif

// ---------------------------------------------------------------------------
// Kernel 1: 1x1-conv QKV projection, fp32 -> bf16, with reference layouts:
//   q,v : [b][n = t*1600 + w*40 + h][c]
//   k   : scrambled reshape baked in: flat L = w*T*H*CS + t*H*CS + h*CS + cc
//         stored at kb[b][m = L%N][c = L/N]  so energy = Q @ Kmat^T
// ---------------------------------------------------------------------------
__global__ void qkv_proj_kernel(const float* __restrict__ x,
                                const float* __restrict__ w_q, const float* __restrict__ b_q,
                                const float* __restrict__ w_k, const float* __restrict__ b_k,
                                const float* __restrict__ w_v, const float* __restrict__ b_v,
                                unsigned short* __restrict__ qb,
                                unsigned short* __restrict__ kb,
                                unsigned short* __restrict__ vb) {
  const int pid   = blockIdx.x * blockDim.x + threadIdx.x;   // 0..25599
  const int which = blockIdx.y;                              // 0=q 1=k 2=v
  const int bt = pid / PP, p = pid % PP;
  const int b  = bt >> 2,  t = bt & 3;
  const int w  = p / HD,   h = p % HD;

  const float* wm = (which == 0) ? w_q : (which == 1) ? w_k : w_v;
  const float* bs = (which == 0) ? b_q : (which == 1) ? b_k : b_v;

  float acc[CS];
#pragma unroll
  for (int o = 0; o < CS; ++o) acc[o] = bs[o];

  const float* xp = x + (size_t)bt * CIN * PP + p;
  for (int c = 0; c < CIN; ++c) {
    float xv = xp[(size_t)c * PP];
#pragma unroll
    for (int o = 0; o < CS; ++o) acc[o] = fmaf(xv, wm[o * CIN + c], acc[o]);
  }

  if (which != 1) {
    const int n = t * PP + p;
    unsigned short* dst = ((which == 0) ? qb : vb) + ((size_t)(b * NN + n)) * CS;
#pragma unroll
    for (int o = 0; o < CS; o += 2)
      *(unsigned int*)(dst + o) = pk2(f2bf(acc[o]), f2bf(acc[o + 1]));
  } else {
    const int Lbase = w * (TT * HD * CS) + t * (HD * CS) + h * CS;
#pragma unroll
    for (int o = 0; o < CS; ++o) {
      int L = Lbase + o;
      int c = L / NN;
      int m = L - c * NN;
      kb[((size_t)(b * NN + m)) * CS + c] = f2bf(acc[o]);
    }
  }
}

// ---------------------------------------------------------------------------
// Kernel 2: flash attention. 8 waves/block, 16 Q-rows per wave, streams 200
// key-tiles of 32. WMMA bf16 for S = Q K^T and O += P V, online softmax fp32
// with DPP16 reductions. P stored key-interleaved (s = 2*(k&15) + (k>>4)) so
// the lane's two probabilities pack into one dword store; V is staged into LDS
// with the same key permutation (sum over k is permutation-invariant).
// V tile is staged with plain global loads and per-element LDS stores.
// ---------------------------------------------------------------------------
__global__ void __launch_bounds__(256) flash_attn_kernel(
    const unsigned short* __restrict__ qb,
    const unsigned short* __restrict__ kb,
    const unsigned short* __restrict__ vb,
    float* __restrict__ attw) {
  __shared__ unsigned short Kt[CS][KT];     // row major:  [key][c]
  __shared__ unsigned short Vt[KT][CS];     // transposed: [c][key slot], keys permuted
  __shared__ unsigned short Pt[8][16][KT];  // per-wave P, key-interleaved

  const int tid  = threadIdx.x;
  const int wave = tid >> 5;
  const int lane = tid & 31;
  const int l15  = lane & 15;
  const int grp  = lane >> 4;
  const int vrow = (lane >> 1) + ((lane & 1) << 4);  // inverse key permutation
  const int b    = blockIdx.y;
  const int row0 = blockIdx.x * 128 + wave * 16;

  // Q tile as A operand (16x32 bf16): lane=M, K-pairs split across lane halves
  AB A;
  {
    const unsigned short* qrow = qb + ((size_t)(b * NN + row0 + l15)) * CS;
#pragma unroll
    for (int v = 0; v < 8; ++v) {
      int c0 = ((v < 4) ? 2 * v : 16 + 2 * (v - 4)) + grp * 8;
      A.u[v] = *(const unsigned int*)(qrow + c0);
    }
  }

  v8f O0 = {0.f,0.f,0.f,0.f,0.f,0.f,0.f,0.f};
  v8f O1 = O0;
  float rm[8], rs[8];
#pragma unroll
  for (int r = 0; r < 8; ++r) { rm[r] = -3.0e38f; rs[r] = 0.0f; }

  const int skey = tid >> 3;        // staging: one key row,
  const int scg  = (tid & 7) * 4;   //          4 channels per thread

  for (int kt = 0; kt < NKT; ++kt) {
    {
      const size_t srcK = ((size_t)(b * NN + kt * KT + skey)) * CS + scg;
      *(uint2*)&Kt[skey][scg] = *(const uint2*)(kb + srcK);
      uint2 dv = *(const uint2*)(vb + srcK);
      const int vs = 2 * (skey & 15) + (skey >> 4);
      Vt[scg + 0][vs] = (unsigned short)(dv.x & 0xffffu);
      Vt[scg + 1][vs] = (unsigned short)(dv.x >> 16);
      Vt[scg + 2][vs] = (unsigned short)(dv.y & 0xffffu);
      Vt[scg + 3][vs] = (unsigned short)(dv.y >> 16);
      if (kt + 1 < NKT) {
        __builtin_prefetch(kb + srcK + (size_t)KT * CS, 0, 1);
        __builtin_prefetch(vb + srcK + (size_t)KT * CS, 0, 1);
      }
    }
    __syncthreads();

    // S = Q * K^T : two 16x16 tiles over this 32-key block
    AB B0, B1;
#pragma unroll
    for (int v = 0; v < 8; ++v) {
      int c0 = ((v < 4) ? 2 * v : 16 + 2 * (v - 4)) + grp * 8;
      B0.u[v] = *(const unsigned int*)&Kt[l15][c0];
      B1.u[v] = *(const unsigned int*)&Kt[16 + l15][c0];
    }
    v8f zc = {0.f,0.f,0.f,0.f,0.f,0.f,0.f,0.f};
    v8f S0 = __builtin_amdgcn_wmma_f32_16x16x32_bf16(false, A.v, false, B0.v, (short)0, zc, false, false);
    v8f S1 = __builtin_amdgcn_wmma_f32_16x16x32_bf16(false, A.v, false, B1.v, (short)0, zc, false, false);

    // online softmax; C layout: VGPR r / lane-half grp -> row r + 8*grp
#pragma unroll
    for (int r = 0; r < 8; ++r) {
      float s0 = S0[r], s1 = S1[r];
      float mnew  = fmaxf(rm[r], rowmax16(fmaxf(s0, s1)));
      float scale = __expf(rm[r] - mnew);
      rm[r] = mnew;
      float p0 = __expf(s0 - mnew);
      float p1 = __expf(s1 - mnew);
      rs[r] = rs[r] * scale + rowsum16(p0 + p1);
      O0[r] *= scale;
      O1[r] *= scale;
      // keys l15 and l15+16 -> interleaved slots 2*l15, 2*l15+1 (one dword)
      *(unsigned int*)&Pt[wave][r + 8 * grp][2 * l15] = pk2(f2bf(p0), f2bf(p1));
    }

    asm volatile("s_wait_dscnt 0x0" ::: "memory");  // intra-wave cross-lane LDS RAW

    // O += P * V : P re-read in A layout; V rows permuted to match P's key order
    AB PA, V0, V1;
#pragma unroll
    for (int v = 0; v < 8; ++v) {
      int c0 = ((v < 4) ? 2 * v : 16 + 2 * (v - 4)) + grp * 8;
      PA.u[v] = *(const unsigned int*)&Pt[wave][l15][c0];
      V0.u[v] = *(const unsigned int*)&Vt[l15][c0];
      V1.u[v] = *(const unsigned int*)&Vt[16 + l15][c0];
    }
    O0 = __builtin_amdgcn_wmma_f32_16x16x32_bf16(false, PA.v, false, V0.v, (short)0, O0, false, false);
    O1 = __builtin_amdgcn_wmma_f32_16x16x32_bf16(false, PA.v, false, V1.v, (short)0, O1, false, false);

    __syncthreads();
  }

#pragma unroll
  for (int r = 0; r < 8; ++r) {
    float inv = 1.0f / rs[r];
    size_t o = ((size_t)(b * NN + row0 + r + 8 * grp)) * CS + l15;
    attw[o]      = O0[r] * inv;
    attw[o + 16] = O1[r] * inv;
  }
}

// ---------------------------------------------------------------------------
// Kernel 3: out = gamma * (attn @ w_o^T + b_o) + x
// ---------------------------------------------------------------------------
__global__ void out_proj_kernel(const float* __restrict__ attw,
                                const float* __restrict__ w_o,
                                const float* __restrict__ b_o,
                                const float* __restrict__ gamma,
                                const float* __restrict__ x,
                                float* __restrict__ out) {
  const int idx = blockIdx.x * blockDim.x + threadIdx.x;  // bt*CIN*PP + o*PP + p
  const int p   = idx % PP;
  const int rem = idx / PP;
  const int o   = rem % CIN;
  const int bt  = rem / CIN;
  const int bb  = bt >> 2, t = bt & 3;
  const float* arow = attw + ((size_t)(bb * NN + t * PP + p)) * CS;
  const float* wrow = w_o + o * CS;
  float acc = b_o[o];
#pragma unroll
  for (int c = 0; c < CS; ++c) acc = fmaf(arow[c], wrow[c], acc);
  out[idx] = gamma[0] * acc + x[idx];
}

extern "C" void kernel_launch(void* const* d_in, const int* in_sizes, int n_in,
                              void* d_out, int out_size, void* d_ws, size_t ws_size,
                              hipStream_t stream) {
  const float* x   = (const float*)d_in[0];
  const float* w_q = (const float*)d_in[1];
  const float* b_q = (const float*)d_in[2];
  const float* w_k = (const float*)d_in[3];
  const float* b_k = (const float*)d_in[4];
  const float* w_v = (const float*)d_in[5];
  const float* b_v = (const float*)d_in[6];
  const float* w_o = (const float*)d_in[7];
  const float* b_o = (const float*)d_in[8];
  const float* gm  = (const float*)d_in[9];

  const size_t QS = (size_t)BB * NN * CS;       // elements per bf16 buffer
  unsigned short* qb = (unsigned short*)d_ws;
  unsigned short* kb = qb + QS;
  unsigned short* vb = kb + QS;
  float* attw = (float*)(vb + QS);              // fp32 attention output

  qkv_proj_kernel<<<dim3((BB * TT * PP) / 256, 3), 256, 0, stream>>>(
      x, w_q, b_q, w_k, b_k, w_v, b_v, qb, kb, vb);
  flash_attn_kernel<<<dim3(NN / 128, BB), 256, 0, stream>>>(qb, kb, vb, attw);
  out_proj_kernel<<<(BB * TT * CIN * PP) / 256, 256, 0, stream>>>(
      attw, w_o, b_o, gm, x, (float*)d_out);
}
